// NGU_46007689674956
// MI455X (gfx1250) — matrix operand order repaired and checked by reading, not compiled, not run on hardware
//
#include <hip/hip_runtime.h>

// ---------------------------------------------------------------------------
// NGU episodic reward for MI455X (gfx1250, wave32, WMMA + TDM).
//
// Roofline: layer-1 GEMM (8192x28224 @ 28224x128) = 59 GFLOP over 924 MB of
// fp32 episode reads -> 63 FLOP/B -> ~40 us HBM floor @ 23.3 TB/s. fp32 rows
// are DMA-staged to LDS by the Tensor Data Mover, converted with
// v_cvt_pk_bf16_f32, and fed to v_wmma_f32_16x16x32_bf16. W1 is transposed /
// converted once to bf16 (7.2 MB, resident in the 192 MB L2).
//
// Parallel shape: one block = one 16-row strip; its 8 waves share the rows
// but each owns a 16x16 output tile (512x8 = 4096 waves + tail), so each
// wave needs only ~48 VGPRs (1 acc, 1 A frag, 2 B frags) -> deep occupancy
// to hide the per-wave load->wmma dependency. Each wave TDMs the strip into
// its private triple-buffered LDS region (7 of 8 fetches hit L2; HBM traffic
// stays 1x) and double-buffers its B stream across k-iterations so the
// global load for chunk i+1 issues before the WMMA of chunk i.
//
// kernel(dist) is monotone in dist, so one top-10-smallest-distances pass
// yields both outputs (reward, new_dm2).
// ---------------------------------------------------------------------------

typedef __attribute__((ext_vector_type(16))) __bf16        v16bf;
typedef __attribute__((ext_vector_type(8)))  float         v8f;
typedef __attribute__((ext_vector_type(4)))  unsigned int  v4u;
typedef __attribute__((ext_vector_type(8)))  int           v8i;
typedef __attribute__((ext_vector_type(4)))  int           v4i;

#define K1        28224
#define N_EP      8192
#define CHUNK     32
#define NCH       (K1 / CHUNK)   // 882
#define DEPTH     3              // TDM pipeline depth (private LDS buffers)
#define EP_BLOCKS 512            // 16-row strips of the episode

#if __has_builtin(__builtin_amdgcn_tensor_load_to_lds) && \
    __has_builtin(__builtin_amdgcn_s_wait_tensorcnt)
#define HAVE_TDM 1
#else
#define HAVE_TDM 0
#endif

union ABu { v16bf v; __bf16 b[16]; unsigned short h[16]; uint4 q[2]; };

__device__ inline unsigned short f2bf(float f) {
  union { __bf16 b; unsigned short u; } c;
  c.b = (__bf16)f;            // v_cvt_*_bf16_f32 (RNE) on gfx1250
  return c.u;
}

// A-fragment (16x32 bf16, ISA layout): lanes l and l+16 share row M=l&15.
// elems 0..7  = K kb+koff+0..7 (koff=0 lanes 0-15, 8 lanes 16-31)
// elems 8..15 = K kb+16+koff+0..7
__device__ inline v16bf cvt_a_frag(const float* p0, const float* p1) {
  float t[16];
  *(float4*)(t + 0)  = *(const float4*)(p0);
  *(float4*)(t + 4)  = *(const float4*)(p0 + 4);
  *(float4*)(t + 8)  = *(const float4*)(p1);
  *(float4*)(t + 12) = *(const float4*)(p1 + 4);
  ABu f;
#pragma unroll
  for (int i = 0; i < 16; ++i) f.b[i] = (__bf16)t[i];
  return f.v;
}

// A-fragment from bf16 LDS (two 16B halves, K gap of 16 between them).
__device__ inline v16bf load_a_bf(const unsigned short* p0,
                                  const unsigned short* p1) {
  ABu f;
  f.q[0] = *(const uint4*)p0;
  f.q[1] = *(const uint4*)p1;
  return f.v;
}

// B-fragment (32x16 bf16): lane n holds col N=n&15; elems 0..7 = K kb+koff+0..7,
// elems 8..15 = K kb+16+koff+0..7 (koff=0 lanes 0-15, 8 lanes 16-31).
__device__ inline v16bf load_b_frag(const unsigned short* p) {
  ABu f;
  f.q[0] = *(const uint4*)(p);
  f.q[1] = *(const uint4*)(p + 16);
  return f.v;
}

__device__ inline v8f wmma_bf16(v16bf a, v16bf b, v8f c) {
  return __builtin_amdgcn_wmma_f32_16x16x32_bf16(false, a, false, b,
                                                 (short)0, c, false, false);
}

#if HAVE_TDM
// One TDM load: 2-D tile [16 rows x CHUNK cols] of fp32 from a row-major
// tensor (row stride K1, `rows` valid rows; OOB rows read zero) into LDS.
// D# per ISA ch.8: g0 = count=1 | lds_addr | global_addr(57b) | type=2;
// g1 = data_size=4B | tensor_dim0=K1 | tensor_dim1=rows | tile=CHUNKx16 |
// tensor_dim0_stride=K1; g2/g3 zero (2-D).
__device__ inline void tdm_issue(unsigned lds_off, const float* gptr,
                                 unsigned rows) {
  unsigned long long ga = (unsigned long long)(size_t)gptr;
  v4u g0 = { 1u,                                   // count=1 (valid user D#)
             lds_off,
             (unsigned)(ga & 0xFFFFFFFFu),
             (unsigned)((ga >> 32) & 0x1FFFFFFu) | (2u << 30) };  // type=2
  v8i g1 = { (int)(2u << 16),                      // data_size: 4 bytes
             (int)((K1 & 0xFFFFu) << 16),          // tensor_dim0[15:0]
             (int)((K1 >> 16) | (rows << 16)),     // dim0[31:16] | dim1[15:0]
             (int)(((unsigned)CHUNK) << 16),       // dim1 hi=0 | tile_dim0
             16,                                   // tile_dim1=16, tile_dim2=0
             (int)K1,                              // tensor_dim0_stride[31:0]
             0, 0 };
  v4i gz = { 0, 0, 0, 0 };
#if defined(__clang_major__) && __clang_major__ >= 23
  v8i gz8 = { 0, 0, 0, 0, 0, 0, 0, 0 };
  __builtin_amdgcn_tensor_load_to_lds(g0, g1, gz, gz, gz8, 0);
#else
  __builtin_amdgcn_tensor_load_to_lds(g0, g1, gz, gz, 0);
#endif
}
#endif

// ---------------------------------------------------------------------------
// One-time weight transpose + fp32->bf16 (column-major Wt[n][k]).
// ---------------------------------------------------------------------------
__global__ __launch_bounds__(256) void prep_weights(
    const float* __restrict__ W1, const float* __restrict__ W2,
    const float* __restrict__ W3, unsigned short* __restrict__ Wt1,
    unsigned short* __restrict__ Wt2, unsigned short* __restrict__ Wt3) {
  long long i = (long long)blockIdx.x * 256 + threadIdx.x;
  const long long n1 = (long long)128 * K1;
  if (i < n1) {
    int n = (int)(i / K1), k = (int)(i % K1);
    Wt1[i] = f2bf(W1[(long long)k * 128 + n]);
    return;
  }
  i -= n1;
  if (i < 64 * 128) {
    int n = (int)(i / 128), k = (int)(i % 128);
    Wt2[i] = f2bf(W2[k * 64 + n]);
    return;
  }
  i -= 64 * 128;
  if (i < 32 * 64) {
    int n = (int)(i / 64), k = (int)(i % 64);
    Wt3[i] = f2bf(W3[k * 32 + n]);
  }
}

// ---------------------------------------------------------------------------
// Fused 3-layer MLP. Grid = EP_BLOCKS+1; block b owns rows 16b..16b+15
// (block 512: the single query row s, tensor_dim1=1 -> OOB rows read zero).
// ---------------------------------------------------------------------------
__global__ __launch_bounds__(256) void embed_kernel(
    const float* __restrict__ s, const float* __restrict__ ep,
    const float* __restrict__ b1, const float* __restrict__ b2,
    const float* __restrict__ b3, const unsigned short* __restrict__ Wt1,
    const unsigned short* __restrict__ Wt2,
    const unsigned short* __restrict__ Wt3, float* __restrict__ Z) {
  __shared__ float At[8][DEPTH][16 * CHUNK];   // 48 KB: per-wave TDM buffers
  __shared__ unsigned short Hb[16 * 128];      // 4 KB: block H1 (bf16)
  __shared__ unsigned short H2b[16 * 64];      // 2 KB: block H2 (bf16)

  const int lane = threadIdx.x & 31;
  const int wave = threadIdx.x >> 5;
  const int lm = lane & 15;
  const int hi = lane >> 4;
  const int hi8 = hi * 8;
  const int koff = hi * 8;     // A-frag K offset
  const int kof16 = hi * 8;    // B-frag K offset
  const int rowBase = blockIdx.x * 16;

  const bool isTail = (blockIdx.x == EP_BLOCKS);
  const float* tileG = isTail ? s : (ep + (size_t)rowBase * K1);
  const unsigned rows = isTail ? 1u : 16u;

#if HAVE_TDM
  tdm_issue((unsigned)(size_t)&At[wave][0][0], tileG, rows);
  tdm_issue((unsigned)(size_t)&At[wave][1][0], tileG + CHUNK, rows);
  tdm_issue((unsigned)(size_t)&At[wave][2][0], tileG + 2 * CHUNK, rows);
#endif

  // ---- layer 1: this wave's 16x16 tile, cols wave*16..wave*16+15
  const unsigned short* bcol = Wt1 + (size_t)(wave * 16 + lm) * K1 + kof16;
  v16bf bcur = load_b_frag(bcol);
  v8f acc = {};
  int cur = 0;
  for (int i = 0; i < NCH; ++i) {
    // next chunk's B stream issues before this chunk's WMMA (2-frag pipeline)
    int nx = (i + 1 < NCH) ? (i + 1) : (NCH - 1);
    v16bf bnext = load_b_frag(bcol + nx * CHUNK);
#if HAVE_TDM
    int rem = NCH - 1 - i;     // chunks still needed after this one
    if (rem >= 2)      __builtin_amdgcn_s_wait_tensorcnt((short)2);
    else if (rem == 1) __builtin_amdgcn_s_wait_tensorcnt((short)1);
    else               __builtin_amdgcn_s_wait_tensorcnt((short)0);
#else
    {  // fallback: per-wave synchronous copy (each lane: 16 floats)
      int r = lane >> 1;
      if ((unsigned)r >= rows) r = (int)rows - 1;
      const float* src = tileG + (size_t)r * K1 + i * CHUNK + (lane & 1) * 16;
      float* dst = &At[wave][cur][(lane >> 1) * CHUNK + (lane & 1) * 16];
      *(float4*)(dst + 0)  = *(const float4*)(src + 0);
      *(float4*)(dst + 4)  = *(const float4*)(src + 4);
      *(float4*)(dst + 8)  = *(const float4*)(src + 8);
      *(float4*)(dst + 12) = *(const float4*)(src + 12);
    }
#endif
    const float* cb = &At[wave][cur][0];
    v16bf a = cvt_a_frag(cb + lm * CHUNK + koff, cb + lm * CHUNK + 16 + koff);
    acc = wmma_bf16(a, bcur, acc);
#if HAVE_TDM
    // buffer `cur` fully consumed above; refill it with chunk i+DEPTH
    if (i + DEPTH < NCH)
      tdm_issue((unsigned)(size_t)&At[wave][cur][0],
                tileG + (i + DEPTH) * CHUNK, rows);
#endif
    bcur = bnext;
    cur = (cur + 1 == DEPTH) ? 0 : cur + 1;
  }

  // bias + relu -> block-shared bf16 H1 (cols wave*16..+15)
  {
    float bv = b1[wave * 16 + lm];
#pragma unroll
    for (int r = 0; r < 8; ++r)
      Hb[(r + hi8) * 128 + wave * 16 + lm] = f2bf(fmaxf(acc[r] + bv, 0.0f));
  }
  __syncthreads();

  // ---- layer 2: [16x128] @ [128x64]; waves 0-3, one 16x16 tile each
  if (wave < 4) {
    v8f acc2 = {};
#pragma unroll
    for (int kc = 0; kc < 128; kc += 32) {
      v16bf b = load_b_frag(Wt2 + (wave * 16 + lm) * 128 + kc + kof16);
      v16bf a = load_a_bf(Hb + lm * 128 + kc + koff,
                          Hb + lm * 128 + kc + 16 + koff);
      acc2 = wmma_bf16(a, b, acc2);
    }
    float bv = b2[wave * 16 + lm];
#pragma unroll
    for (int r = 0; r < 8; ++r)
      H2b[(r + hi8) * 64 + wave * 16 + lm] = f2bf(fmaxf(acc2[r] + bv, 0.0f));
  }
  __syncthreads();

  // ---- layer 3: [16x64] @ [64x32]; waves 0-1
  if (wave < 2) {
    v8f acc3 = {};
#pragma unroll
    for (int kc = 0; kc < 64; kc += 32) {
      v16bf b = load_b_frag(Wt3 + (wave * 16 + lm) * 64 + kc + kof16);
      v16bf a = load_a_bf(H2b + lm * 64 + kc + koff,
                          H2b + lm * 64 + kc + 16 + koff);
      acc3 = wmma_bf16(a, b, acc3);
    }
    float bv = b3[wave * 16 + lm];
#pragma unroll
    for (int r = 0; r < 8; ++r)
      Z[(size_t)(rowBase + r + hi8) * 32 + wave * 16 + lm] = acc3[r] + bv;
  }
}

// ---------------------------------------------------------------------------
// dist[n] = || Z[n] - Z[8192] ||_2
// ---------------------------------------------------------------------------
__global__ __launch_bounds__(256) void dist_kernel(const float* __restrict__ Z,
                                                   float* __restrict__ dist) {
  int n = blockIdx.x * 256 + threadIdx.x;
  if (n >= N_EP) return;
  const float* zs = Z + (size_t)N_EP * 32;
  const float* zn = Z + (size_t)n * 32;
  float sum = 0.0f;
#pragma unroll
  for (int f = 0; f < 32; ++f) {
    float d = zn[f] - zs[f];
    sum += d * d;
  }
  dist[n] = sqrtf(sum);
}

// ---------------------------------------------------------------------------
// Single-block top-10 smallest distances -> both scalars.
// ---------------------------------------------------------------------------
__global__ __launch_bounds__(256) void topk_kernel(
    const float* __restrict__ dist, const float* __restrict__ dm2p,
    float* __restrict__ out) {
  __shared__ float sd[N_EP];
  __shared__ float rv[256];
  __shared__ int ri[256];
  const int tid = threadIdx.x;
  for (int i = tid; i < N_EP; i += 256) sd[i] = dist[i];
  __syncthreads();

  const float dm2 = dm2p[0];
  float sumD = 0.0f, sumK = 0.0f;  // valid on tid 0 only
  for (int it = 0; it < 10; ++it) {
    float best = 3.4e38f;
    int bi = 0;
    for (int i = tid; i < N_EP; i += 256) {
      float v = sd[i];
      if (v < best) { best = v; bi = i; }
    }
    rv[tid] = best;
    ri[tid] = bi;
    __syncthreads();
    for (int s2 = 128; s2 > 0; s2 >>= 1) {
      if (tid < s2 && rv[tid + s2] < rv[tid]) {
        rv[tid] = rv[tid + s2];
        ri[tid] = ri[tid + s2];
      }
      __syncthreads();
    }
    if (tid == 0) {
      float d = rv[0];
      sumD += d;
      sumK += 0.001f / (d / dm2 + 0.001f);
      sd[ri[0]] = 3.4e38f;
    }
    __syncthreads();
  }
  if (tid == 0) {
    float meanD = sumD * 0.1f;
    float meanK = sumK * 0.1f;
    out[0] = 1.0f / (sqrtf(meanK) + 0.001f);  // reward_episodic
    out[1] = 0.99f * dm2 + 0.01f * meanD;     // new_dm2
  }
}

// ---------------------------------------------------------------------------
// Workspace layout (bytes, 256-aligned):
//   Wt1 bf16 128x28224 : [0, 7225344)
//   Wt2 bf16  64x128   : [7225344, 7241728)
//   Wt3 bf16  32x64    : [7241728, 7245824)
//   Z   f32 8320x32    : [7245824, 8310784)   (8208 rows used)
//   dist f32 8192      : [8310784, 8343552)
// ---------------------------------------------------------------------------
extern "C" void kernel_launch(void* const* d_in, const int* in_sizes, int n_in,
                              void* d_out, int out_size, void* d_ws,
                              size_t ws_size, hipStream_t stream) {
  const float* s   = (const float*)d_in[0];
  const float* ep  = (const float*)d_in[1];
  const float* dm2 = (const float*)d_in[2];
  const float* W1  = (const float*)d_in[3];
  const float* b1  = (const float*)d_in[4];
  const float* W2  = (const float*)d_in[5];
  const float* b2  = (const float*)d_in[6];
  const float* W3  = (const float*)d_in[7];
  const float* b3  = (const float*)d_in[8];
  float* out = (float*)d_out;

  char* ws = (char*)d_ws;
  unsigned short* Wt1 = (unsigned short*)(ws);
  unsigned short* Wt2 = (unsigned short*)(ws + 7225344);
  unsigned short* Wt3 = (unsigned short*)(ws + 7241728);
  float* Z            = (float*)(ws + 7245824);
  float* dist         = (float*)(ws + 8310784);

  prep_weights<<<14152, 256, 0, stream>>>(W1, W2, W3, Wt1, Wt2, Wt3);
  embed_kernel<<<EP_BLOCKS + 1, 256, 0, stream>>>(s, ep, b1, b2, b3, Wt1, Wt2,
                                                  Wt3, Z);
  dist_kernel<<<32, 256, 0, stream>>>(Z, dist);
  topk_kernel<<<1, 256, 0, stream>>>(dist, dm2, out);
}
